// MultiZoomSelfAttention_274877907669
// MI455X (gfx1250) — matrix operand order repaired and checked by reading, not compiled, not run on hardware
//
#include <hip/hip_runtime.h>
#include <hip/hip_bf16.h>
#include <cstdint>

// Problem constants (from reference): B=1 V=2 T=2 S=16384 C=128 NH=9 H=4 DH=32 MULT=2
#define S_    16384
#define C_    128
#define C2_   256
#define NH_   9
#define H_    4
#define DH_   32
#define NTOK  65536      // B*V*T*S

#if defined(__AMDGCN__) && __has_builtin(__builtin_amdgcn_global_load_async_to_lds_b128)
#define USE_ASYNC 1
#else
#define USE_ASYNC 0
#endif

typedef __attribute__((ext_vector_type(16))) __bf16 v16bf;
typedef __attribute__((ext_vector_type(8)))  float  v8f;

static __device__ __forceinline__ v8f wmma_bf16(v16bf a, v16bf b, v8f c) {
  // v_wmma_f32_16x16x32_bf16  (8-arg form: neg_a, A, neg_b, B, c_mod, C, reuse_a, reuse_b)
  return __builtin_amdgcn_wmma_f32_16x16x32_bf16(false, a, false, b, (short)0, c, false, false);
}

// A fragment: 16x32 bf16 from row-major LDS tile, leading dim ldA (elements).
// ISA layout: lanes 0-15: M=lane, K = kb..kb+7 then 16+kb..; kb = (lane>>4)*8.
// Two 16B chunks per lane -> ds_load_b128 x2.
static __device__ __forceinline__ v16bf load_A(const __bf16* At, int ldA, int kofs, int lane) {
  const int row = lane & 15;
  const int kb  = (lane >> 4) << 3;
  const __bf16* p = At + (size_t)row * ldA + kofs + kb;
  union { v16bf v; uint4 q[2]; } u;
  u.q[0] = *(const uint4*)p;          // elements 0..7   (K = kb .. kb+7)
  u.q[1] = *(const uint4*)(p + 16);   // elements 8..15  (K = 16+kb .. 16+kb+7)
  return u.v;
}

// B fragment from pre-packed fragment-major weights: frag = nt*ktiles + kt,
// each fragment is 512 bf16 laid out [lane][16] contiguous -> global_load_b128 x2.
static __device__ __forceinline__ v16bf load_Bp(const __bf16* Wf, int ktiles, int kt, int nt, int lane) {
  const __bf16* p = Wf + ((size_t)(nt * ktiles + kt) << 9) + (lane << 4);
  union { v16bf v; uint4 q[2]; } u;
  u.q[0] = *(const uint4*)p;
  u.q[1] = *(const uint4*)(p + 8);
  return u.v;
}

static __device__ __forceinline__ float wred(float v) {
#pragma unroll
  for (int m = 16; m >= 1; m >>= 1) v += __shfl_xor(v, m, 32);
  return v;
}

// ---------------------------------------------------------------- kernel 0: pack weight into WMMA B-fragment layout
// For W[K][N] row-major: value (k,n) -> frag (nt*Kt + kt), lane = ((k&8)<<1)|(n&15), elem j = (k&7)|((k&16)>>1).
__global__ void k_pack_w(const float* __restrict__ W, __bf16* __restrict__ out, int K, int N) {
  const int i = blockIdx.x * 256 + threadIdx.x;
  if (i >= K * N) return;
  const int k = i / N, n = i % N;
  const int kt = k >> 5, kk = k & 31;
  const int nt = n >> 4, col = n & 15;
  const int lane = ((kk & 8) << 1) | col;
  const int j = (kk & 7) | ((kk & 16) >> 1);
  const int ktiles = K >> 5;
  out[(((size_t)(nt * ktiles + kt)) << 9) + (lane << 4) + j] = (__bf16)W[(size_t)k * N + n];
}

// ---------------------------------------------------------------- kernel 1: LN1 -> xe (bf16)
__global__ void k_ln1(const float* __restrict__ x, const float* __restrict__ g,
                      const float* __restrict__ b, __bf16* __restrict__ xe) {
  const int wave = threadIdx.x >> 5, lane = threadIdx.x & 31;
  const int tok = blockIdx.x * 4 + wave;
  const float4 v = ((const float4*)(x + (size_t)tok * C_))[lane];
  float s  = v.x + v.y + v.z + v.w;
  float ss = v.x * v.x + v.y * v.y + v.z * v.z + v.w * v.w;
  s = wred(s); ss = wred(ss);
  const float mean = s * (1.0f / C_);
  const float rstd = rsqrtf(ss * (1.0f / C_) - mean * mean + 1e-5f);
  const float4 gv = ((const float4*)g)[lane];
  const float4 bv = ((const float4*)b)[lane];
  __bf16* o = xe + (size_t)tok * C_ + lane * 4;
  o[0] = (__bf16)((v.x - mean) * rstd * gv.x + bv.x);
  o[1] = (__bf16)((v.y - mean) * rstd * gv.y + bv.y);
  o[2] = (__bf16)((v.z - mean) * rstd * gv.z + bv.z);
  o[3] = (__bf16)((v.w - mean) * rstd * gv.w + bv.w);
}

// ---------------------------------------------------------------- kernel 2: fused attention + MLP
__global__ void __launch_bounds__(128, 1)
k_attn_mlp(const float* __restrict__ x,
           const int* __restrict__ adjc,
           const unsigned char* __restrict__ mask,
           const float* __restrict__ bkv,
           const float* __restrict__ Wnh,
           const float* __restrict__ ln2g, const float* __restrict__ ln2b,
           const float* __restrict__ b1,   const float* __restrict__ b2,
           const float* __restrict__ gamma,
           const __bf16* __restrict__ xe,
           const __bf16* __restrict__ Wq_b,  const __bf16* __restrict__ Wkv_b,
           const __bf16* __restrict__ Wao_b, const __bf16* __restrict__ W1_b,
           const __bf16* __restrict__ W2_b,  const __bf16* __restrict__ Wout_b,
           float* __restrict__ out)
{
  __shared__ __bf16 sXE[16][C_];          //  4 KB  LN1 rows of this tile
  __shared__ float  sQ[16][C_];           //  8 KB  q
  __shared__ __bf16 sKVG[NH_][16][C_];    // 36 KB  gathered neighbor xe rows
  __shared__ __bf16 sKV[NH_][16][C2_];    // 72 KB  kv (k|v) after bias + mixing
  __shared__ float  sX1[16][C_];          //  8 KB  x + attn out
  __shared__ __bf16 sTMP[16][C2_];        //  8 KB  o / xm / x2 staging
  __shared__ __bf16 sH[16][C2_];          //  8 KB  gelu hidden
  __shared__ float  sWnh[NH_ * NH_];
  __shared__ int    sAdj[NH_ * 16];       // adjc rows for this tile, [n*16 + t]

  const int tid  = threadIdx.x;
  const int wave = tid >> 5, lane = tid & 31;
  const int tok0 = blockIdx.x * 16;
  const int slab = tok0 / S_;             // which (b,v,t) slab; tiles never straddle slabs
  const int s0   = tok0 - slab * S_;
  const size_t slabBase = (size_t)slab * S_;

  if (tid < NH_ * NH_) sWnh[tid] = Wnh[tid];
  if (tid < NH_ * 16) {
    const int n = tid >> 4, t = tid & 15;
    sAdj[tid] = adjc[(s0 + t) * NH_ + n];
  }
  __syncthreads();

#if USE_ASYNC
  // ---- async DMA: xe tile (256 x 16B chunks) + neighbor gather (2304 x 16B chunks)
  {
    typedef int v4i_vs __attribute__((vector_size(16)));
    typedef __attribute__((address_space(1))) v4i_vs* gp128;
    typedef __attribute__((address_space(3))) v4i_vs* lp128;
    for (int idx = tid; idx < 256; idx += 128) {
      const __bf16* src = xe + (size_t)tok0 * C_ + idx * 8;
      __builtin_amdgcn_global_load_async_to_lds_b128(
          (gp128)src, (lp128)(&sXE[0][0] + idx * 8), 0, 0);
    }
    for (int idx = tid; idx < NH_ * 16 * 16; idx += 128) {
      const int r = idx >> 4, part = idx & 15;
      const int n = r >> 4, t = r & 15;
      const __bf16* src = xe + (slabBase + sAdj[r]) * (size_t)C_ + part * 8;
      __builtin_amdgcn_global_load_async_to_lds_b128(
          (gp128)src, (lp128)(&sKVG[n][t][part * 8]), 0, 0);
    }
#if __has_builtin(__builtin_amdgcn_s_wait_asynccnt)
    __builtin_amdgcn_s_wait_asynccnt(0);
#else
    asm volatile("s_wait_asynccnt 0" ::: "memory");
#endif
  }
#else
  // ---- fallback: plain vector copies through VGPRs
  {
    const uint4* s = (const uint4*)(xe + (size_t)tok0 * C_);
    uint4* d = (uint4*)&sXE[0][0];
    for (int i = tid; i < 16 * C_ / 8; i += 128) d[i] = s[i];
  }
  for (int r = tid; r < NH_ * 16; r += 128) {
    const int n = r >> 4, t = r & 15;
    const uint4* s = (const uint4*)(xe + (slabBase + sAdj[r]) * (size_t)C_);
    __builtin_prefetch(s, 0, 0);
    uint4* d = (uint4*)&sKVG[n][t][0];
#pragma unroll
    for (int i = 0; i < C_ / 8; ++i) d[i] = s[i];
  }
#endif
  __syncthreads();

  // ---- q = xe @ Wq   (16x128x128)
  for (int nt = wave; nt < C_ / 16; nt += 4) {
    v8f acc = {};
#pragma unroll
    for (int kt = 0; kt < 4; ++kt)
      acc = wmma_bf16(load_A(&sXE[0][0], C_, kt * 32, lane), load_Bp(Wq_b, 4, kt, nt, lane), acc);
    const int col = lane & 15, rb = (lane >> 4) * 8;
#pragma unroll
    for (int r = 0; r < 8; ++r) sQ[rb + r][nt * 16 + col] = acc[r];
  }

  // ---- kv = gather @ Wkv + b_kv   (9 tiles of 16x128x256)
  for (int job = wave; job < NH_ * (C2_ / 16); job += 4) {
    const int n = job >> 4, nt = job & 15;
    v8f acc = {};
#pragma unroll
    for (int kt = 0; kt < 4; ++kt)
      acc = wmma_bf16(load_A(&sKVG[n][0][0], C_, kt * 32, lane), load_Bp(Wkv_b, 4, kt, nt, lane), acc);
    const int col = lane & 15, rb = (lane >> 4) * 8;
    const float bias = bkv[nt * 16 + col];
#pragma unroll
    for (int r = 0; r < 8; ++r) sKV[n][rb + r][nt * 16 + col] = (__bf16)(acc[r] + bias);
  }
  __syncthreads();

  // ---- neighbor mixing: kv[m] += sum_n kv[n] * Wnh[n][m]   (per (token,d) column)
  for (int idx = tid; idx < 16 * C2_; idx += 128) {
    const int t = idx >> 8, d = idx & 255;
    float v[NH_], o_[NH_];
#pragma unroll
    for (int n = 0; n < NH_; ++n) v[n] = (float)sKV[n][t][d];
#pragma unroll
    for (int m = 0; m < NH_; ++m) {
      float a = v[m];
#pragma unroll
      for (int n = 0; n < NH_; ++n) a += v[n] * sWnh[n * NH_ + m];
      o_[m] = a;
    }
#pragma unroll
    for (int m = 0; m < NH_; ++m) sKV[m][t][d] = (__bf16)o_[m];
  }
  __syncthreads();

  // ---- tiny attention: 16 tokens x 4 heads, 9 keys of DH=32 (waves 0,1 only; wave-uniform branch)
  if (tid < 64) {
    const int t = tid >> 2, h = tid & 3;
    float qv[DH_];
#pragma unroll
    for (int j = 0; j < DH_; ++j) qv[j] = sQ[t][h * DH_ + j];
    float sc[NH_];
    float mx = -1e30f;
    for (int n = 0; n < NH_; ++n) {
      float a = 0.f;
#pragma unroll
      for (int j = 0; j < DH_; ++j) a += qv[j] * (float)sKV[n][t][h * DH_ + j];
      a *= 0.17677669529663687f;                       // 1/sqrt(32)
      if (mask[slabBase + sAdj[n * 16 + t]]) a = -1e9f;
      sc[n] = a;
      mx = fmaxf(mx, a);
    }
    float den = 0.f;
    for (int n = 0; n < NH_; ++n) { sc[n] = __expf(sc[n] - mx); den += sc[n]; }
    const float inv = 1.f / den;
#pragma unroll
    for (int j = 0; j < DH_; ++j) {
      float a = 0.f;
      for (int n = 0; n < NH_; ++n) a += sc[n] * (float)sKV[n][t][C_ + h * DH_ + j];
      sTMP[t][h * DH_ + j] = (__bf16)(a * inv);
    }
  }
  __syncthreads();

  // ---- x1 = x + o @ W_ao
  for (int nt = wave; nt < C_ / 16; nt += 4) {
    v8f acc = {};
#pragma unroll
    for (int kt = 0; kt < 4; ++kt)
      acc = wmma_bf16(load_A(&sTMP[0][0], C2_, kt * 32, lane), load_Bp(Wao_b, 4, kt, nt, lane), acc);
    const int col = lane & 15, rb = (lane >> 4) * 8;
#pragma unroll
    for (int r = 0; r < 8; ++r) {
      const int row = rb + r, cc = nt * 16 + col;
      sX1[row][cc] = acc[r] + x[(size_t)(tok0 + row) * C_ + cc];
    }
  }
  __syncthreads();

  // ---- LN2 -> xm (bf16 in sTMP)
  for (int t = wave * 4; t < wave * 4 + 4; ++t) {
    const int c = lane * 4;
    const float v0 = sX1[t][c], v1 = sX1[t][c + 1], v2 = sX1[t][c + 2], v3 = sX1[t][c + 3];
    float s  = v0 + v1 + v2 + v3;
    float ss = v0 * v0 + v1 * v1 + v2 * v2 + v3 * v3;
    s = wred(s); ss = wred(ss);
    const float mean = s * (1.0f / C_);
    const float rstd = rsqrtf(ss * (1.0f / C_) - mean * mean + 1e-5f);
    sTMP[t][c]     = (__bf16)((v0 - mean) * rstd * ln2g[c]     + ln2b[c]);
    sTMP[t][c + 1] = (__bf16)((v1 - mean) * rstd * ln2g[c + 1] + ln2b[c + 1]);
    sTMP[t][c + 2] = (__bf16)((v2 - mean) * rstd * ln2g[c + 2] + ln2b[c + 2]);
    sTMP[t][c + 3] = (__bf16)((v3 - mean) * rstd * ln2g[c + 3] + ln2b[c + 3]);
  }
  __syncthreads();

  // ---- h = gelu(xm @ W1 + b1)   (16x128x256)
  for (int nt = wave; nt < C2_ / 16; nt += 4) {
    v8f acc = {};
#pragma unroll
    for (int kt = 0; kt < 4; ++kt)
      acc = wmma_bf16(load_A(&sTMP[0][0], C2_, kt * 32, lane), load_Bp(W1_b, 4, kt, nt, lane), acc);
    const int col = lane & 15, rb = (lane >> 4) * 8;
    const float bb = b1[nt * 16 + col];
#pragma unroll
    for (int r = 0; r < 8; ++r) {
      const float u = acc[r] + bb;
      const float gl = 0.5f * u * (1.f + tanhf(0.7978845608028654f * (u + 0.044715f * u * u * u)));
      sH[rb + r][nt * 16 + col] = (__bf16)gl;
    }
  }
  __syncthreads();

  // ---- x2 = x1 + gamma * (h @ W2 + b2)   (16x256x128)
  for (int nt = wave; nt < C_ / 16; nt += 4) {
    v8f acc = {};
#pragma unroll
    for (int kt = 0; kt < 8; ++kt)
      acc = wmma_bf16(load_A(&sH[0][0], C2_, kt * 32, lane), load_Bp(W2_b, 8, kt, nt, lane), acc);
    const int col = lane & 15, rb = (lane >> 4) * 8;
    const int cc = nt * 16 + col;
    const float gm = gamma[cc], bb = b2[cc];
#pragma unroll
    for (int r = 0; r < 8; ++r)
      sTMP[rb + r][cc] = (__bf16)(sX1[rb + r][cc] + gm * (acc[r] + bb));
  }
  __syncthreads();

  // ---- out = x2 @ Wout
  for (int nt = wave; nt < C_ / 16; nt += 4) {
    v8f acc = {};
#pragma unroll
    for (int kt = 0; kt < 4; ++kt)
      acc = wmma_bf16(load_A(&sTMP[0][0], C2_, kt * 32, lane), load_Bp(Wout_b, 4, kt, nt, lane), acc);
    const int col = lane & 15, rb = (lane >> 4) * 8;
#pragma unroll
    for (int r = 0; r < 8; ++r)
      out[(size_t)(tok0 + rb + r) * C_ + nt * 16 + col] = acc[r];
  }
}

// ---------------------------------------------------------------- launch
extern "C" void kernel_launch(void* const* d_in, const int* in_sizes, int n_in,
                              void* d_out, int out_size, void* d_ws, size_t ws_size,
                              hipStream_t stream) {
  const float* x     = (const float*)d_in[0];
  const int*   adjc  = (const int*)d_in[1];
  const unsigned char* mask = (const unsigned char*)d_in[2];   // jnp bool_
  const float* ln1g  = (const float*)d_in[3];
  const float* ln1b  = (const float*)d_in[4];
  const float* Wq    = (const float*)d_in[5];
  const float* Wkv   = (const float*)d_in[6];
  const float* bkv   = (const float*)d_in[7];
  const float* Wnh   = (const float*)d_in[8];
  const float* Wao   = (const float*)d_in[9];
  const float* ln2g  = (const float*)d_in[10];
  const float* ln2b  = (const float*)d_in[11];
  const float* W1    = (const float*)d_in[12];
  const float* b1    = (const float*)d_in[13];
  const float* W2    = (const float*)d_in[14];
  const float* b2    = (const float*)d_in[15];
  const float* gamma = (const float*)d_in[16];
  const float* Wout  = (const float*)d_in[17];
  float* out = (float*)d_out;

  // workspace layout: [xe bf16: NTOK*128][packed weights bf16: 147456]  (~17.1 MB)
  __bf16* xe = (__bf16*)d_ws;
  __bf16* wb = (__bf16*)((char*)d_ws + (size_t)NTOK * C_ * 2);
  __bf16* Wq_b   = wb;
  __bf16* Wkv_b  = wb + 16384;
  __bf16* Wao_b  = wb + 49152;
  __bf16* W1_b   = wb + 65536;
  __bf16* W2_b   = wb + 98304;
  __bf16* Wout_b = wb + 131072;

  k_pack_w<<<64,  256, 0, stream>>>(Wq,   Wq_b,   C_,  C_);
  k_pack_w<<<128, 256, 0, stream>>>(Wkv,  Wkv_b,  C_,  C2_);
  k_pack_w<<<64,  256, 0, stream>>>(Wao,  Wao_b,  C_,  C_);
  k_pack_w<<<128, 256, 0, stream>>>(W1,   W1_b,   C_,  C2_);
  k_pack_w<<<128, 256, 0, stream>>>(W2,   W2_b,   C2_, C_);
  k_pack_w<<<64,  256, 0, stream>>>(Wout, Wout_b, C_,  C_);

  k_ln1<<<NTOK / 4, 128, 0, stream>>>(x, ln1g, ln1b, xe);

  k_attn_mlp<<<NTOK / 16, 128, 0, stream>>>(x, adjc, mask, bkv, Wnh, ln2g, ln2b, b1, b2, gamma,
                                            xe, Wq_b, Wkv_b, Wao_b, W1_b, W2_b, Wout_b, out);
}
